// FlashSVDBlock_6786048327887
// MI455X (gfx1250) — compile-verified, matrix operand order checked by reading, not run on hardware
//
#include <hip/hip_runtime.h>
#include <hip/hip_bf16.h>

// ---------------------------------------------------------------------------
// FlashSVD transformer block for MI455X (gfx1250, wave32, WMMA, TDM).
// - All GEMMs on v_wmma_f32_16x16x32_bf16 (f32 accumulate).
// - Weights pre-converted to bf16 and pre-transposed to [N][K] once, so all
//   GEMM tile staging is raw bf16 -> double-buffered TDM DMA with hardware
//   LDS padding (no per-element VALU/VMEM in the GEMM inner loop).
// - Flash attention: scores never touch HBM; Q/K tiles TDM-staged.
// ---------------------------------------------------------------------------

typedef __attribute__((ext_vector_type(16))) __bf16     v16bf;
typedef __attribute__((ext_vector_type(8)))  float      v8f;
typedef __attribute__((ext_vector_type(8)))  unsigned   v8u;

constexpr int B_   = 8;
constexpr int M_   = 1024;
constexpr int DM_  = 1024;
constexpr int H_   = 16;
constexpr int DH_  = 64;
constexpr int R_   = 32;
constexpr int DFF_ = 4096;
constexpr int RF_  = 512;          // RO = R1 = R2 = 512
constexpr int BM_  = B_ * M_;      // 8192 rows

// LDS strides (elements). 40/72 keep rows 16B-aligned (80B / 144B).
#define LDA_S 40   // 64x32 tiles (A and transposed-B): TDM pad 16DW+4DW
#define QS_LD 72   // 64x64 bf16 tiles in attention:    TDM pad 32DW+4DW
#define SS_LD 68   // 64x64 f32 score tile

// ------------------------- TDM (Tensor Data Mover) --------------------------
#if defined(__HIP_DEVICE_COMPILE__) && __has_builtin(__builtin_amdgcn_tensor_load_to_lds)
#define USE_TDM 1
#endif

typedef __attribute__((ext_vector_type(4))) unsigned tdm_ui4;
typedef __attribute__((ext_vector_type(8))) int      tdm_i8;
typedef __attribute__((ext_vector_type(4))) int      tdm_i4;

// DMA a tile_w x tile_h bf16 tile (row stride row_stride elems in memory)
// into LDS; pad_interval/pad_amount codes make the TDM pad engine emit the
// padded LDS row stride (pi=3: every 16 DWORDs, pi=4: every 32 DWORDs;
// pa=3: add 4 DWORDs). data_size code 1 = 2 bytes.
__device__ inline void tdm_load_2d(const __bf16* gsrc, const __bf16* lds,
                                   int tile_w, int tile_h, int row_stride,
                                   unsigned pi, unsigned pa) {
#ifdef USE_TDM
  const unsigned long long ga = (unsigned long long)(size_t)gsrc;
  const unsigned lb = (unsigned)(size_t)lds;  // generic LDS addr: low 32 = offset
  tdm_ui4 g0 = { 1u /*count*/, lb, (unsigned)ga,
                 (unsigned)((ga >> 32) & 0x01FFFFFFu) | (2u << 30) /*type=2*/ };
  const unsigned td0 = (unsigned)row_stride;   // tensor_dim0 >= tile_w
  const unsigned td1 = 1u << 20;               // tensor_dim1: large
  tdm_i8 g1 = { (int)((1u << 16) /*2B*/ | (1u << 20) /*pad_en*/ |
                      (pi << 22) | (pa << 25)),
                (int)((td0 & 0xFFFFu) << 16),
                (int)(((td0 >> 16) & 0xFFFFu) | ((td1 & 0xFFFFu) << 16)),
                (int)(((td1 >> 16) & 0xFFFFu) | ((unsigned)tile_w << 16)),
                (int)tile_h, (int)row_stride, 0, 0 };
  tdm_i4 gz = { 0, 0, 0, 0 };
#if __has_include(<hip/amd_detail/amd_gfx1250_TDM.h>)
  tdm_i8 gz8 = { 0, 0, 0, 0, 0, 0, 0, 0 };
  __builtin_amdgcn_tensor_load_to_lds(g0, g1, gz, gz, gz8, 0);
#else
  __builtin_amdgcn_tensor_load_to_lds(g0, g1, gz, gz, 0);
#endif
#else
  (void)gsrc; (void)lds; (void)tile_w; (void)tile_h; (void)row_stride; (void)pi; (void)pa;
#endif
}

__device__ inline void tdm_wait0() {
#ifdef USE_TDM
#if __has_builtin(__builtin_amdgcn_s_wait_tensorcnt)
  __builtin_amdgcn_s_wait_tensorcnt(0);
#else
  asm volatile("s_wait_tensorcnt 0x0" ::: "memory");
#endif
#endif
}
__device__ inline void tdm_wait2() {
#ifdef USE_TDM
#if __has_builtin(__builtin_amdgcn_s_wait_tensorcnt)
  __builtin_amdgcn_s_wait_tensorcnt(2);
#else
  asm volatile("s_wait_tensorcnt 0x2" ::: "memory");
#endif
#endif
}

// --------------------------- WMMA helpers ----------------------------------

__device__ inline v8f wmma_bf16(v16bf a, v16bf b, v8f c) {
  return __builtin_amdgcn_wmma_f32_16x16x32_bf16(false, a, false, b,
                                                 (short)0, c, false, false);
}

// 16x32 fragment, row-major in LDS, stride ld (even). Serves as A-fragment,
// and as B-fragment when the B tile is staged n-major (transposed).
__device__ inline v16bf load_frag_a(const __bf16* base, int ld) {
  const int lane = threadIdx.x & 31;
  const int row  = lane & 15;
  const int kh   = lane >> 4;
  const unsigned* p = (const unsigned*)(base + row * ld);
  v8u t;
#pragma unroll
  for (int i = 0; i < 4; ++i) {
    t[i]     = p[kh * 4 + i];
    t[i + 4] = p[8 + kh * 4 + i];
  }
  return __builtin_bit_cast(v16bf, t);
}

__device__ inline float gelu_exact(float x) {
  return 0.5f * x * (1.0f + erff(x * 0.70710678118654752f));
}

__device__ inline __bf16 u16_as_bf16(unsigned short u) {
  return __builtin_bit_cast(__bf16, u);
}

// ----------------------- One-time conversion kernels ------------------------

// f32 -> bf16, 4 elements/thread (n multiple of 1024).
__global__ __launch_bounds__(256) void convert_kernel(
    const float* __restrict__ in, __bf16* __restrict__ out) {
  const size_t i = ((size_t)blockIdx.x * 256 + threadIdx.x) * 4;
#pragma unroll
  for (int j = 0; j < 4; ++j) out[i + j] = (__bf16)in[i + j];
}

// in: K x N f32 (batched) -> out: N x K bf16 (transposed), tiled via LDS.
__global__ __launch_bounds__(256) void transpose_convert_kernel(
    const float* __restrict__ in, __bf16* __restrict__ out,
    int K, int N, long ibs, long obs) {
  const float* ib = in + (size_t)blockIdx.z * ibs;
  __bf16*      ob = out + (size_t)blockIdx.z * obs;
  const int n0 = blockIdx.x * 32, k0 = blockIdx.y * 32;
  __shared__ float t[32][33];
  const int tid = threadIdx.x;
#pragma unroll
  for (int j = 0; j < 4; ++j) {
    const int idx = tid + j * 256;
    const int rr = idx >> 5, cc = idx & 31;
    t[rr][cc] = ib[(size_t)(k0 + rr) * N + n0 + cc];
  }
  __syncthreads();
#pragma unroll
  for (int j = 0; j < 4; ++j) {
    const int idx = tid + j * 256;
    const int rr = idx >> 5, cc = idx & 31;
    ob[(size_t)(n0 + rr) * K + k0 + cc] = (__bf16)t[cc][rr];
  }
}

// --------------------------- QKV kernel -------------------------------------
// tmp(64x32) = xb_tile @ P[h] (TDM-pipelined K-loop over DM);
// out(64x64) = tmp @ Vw[h] + bias (single WMMA, R=32).
// Inputs pre-converted: xb bf16 (B,M,DM); PT bf16 (H,R,DM); VwT bf16 (H,DH,R).
// Q pre-scaled by 1/sqrt(DH)=0.125 (exact). Output bf16 (B,H,M,DH).
__global__ __launch_bounds__(256) void qkv_kernel(
    const __bf16* __restrict__ xb,
    const __bf16* __restrict__ PTq, const __bf16* __restrict__ PTk, const __bf16* __restrict__ PTv,
    const __bf16* __restrict__ WTq, const __bf16* __restrict__ WTk, const __bf16* __restrict__ WTv,
    const float* __restrict__ bq, const float* __restrict__ bk, const float* __restrict__ bv,
    __bf16* __restrict__ Qo, __bf16* __restrict__ Ko, __bf16* __restrict__ Vo) {
  const int mt = blockIdx.x;
  const int bh = blockIdx.y;
  const int h  = bh % H_;
  const int z  = blockIdx.z;            // 0=q, 1=k, 2=v

  const __bf16* PT  = (z == 0) ? PTq : (z == 1) ? PTk : PTv;
  const __bf16* WT  = (z == 0) ? WTq : (z == 1) ? WTk : WTv;
  const float*  bs  = (z == 0) ? bq  : (z == 1) ? bk  : bv;
  __bf16*       Out = (z == 0) ? Qo  : (z == 1) ? Ko  : Vo;
  const float   qscale = (z == 0) ? 0.125f : 1.0f;

  __shared__ __attribute__((aligned(16))) __bf16 As[2][64 * LDA_S];
  __shared__ __attribute__((aligned(16))) __bf16 Bs[2][64 * LDA_S];
  __shared__ __attribute__((aligned(16))) __bf16 Ts[64 * LDA_S];

  const int tid = threadIdx.x, wave = tid >> 5, lane = tid & 31;
  const int m0  = mt * 64;
  const __bf16* xblk = xb + ((size_t)(bh / H_) * M_ + m0) * DM_;
  const __bf16* PTh  = PT + (size_t)h * R_ * DM_;
  const __bf16* WTh  = WT + (size_t)h * DH_ * R_;

  // Stage 1: tmp = x @ P ; 8 tiles of 16x16, one per wave; TDM double-buffer.
  const int mi1 = wave >> 1, ni1 = wave & 1;
  v8f acc = {};
#ifdef USE_TDM
  if (wave == 0) {
    tdm_load_2d(xblk, As[0], 32, 64, DM_, 3, 3);
    tdm_load_2d(PTh,  Bs[0], 32, 32, DM_, 3, 3);
  }
#endif
  int p = 0;
  for (int k0 = 0; k0 < DM_; k0 += 32) {
    const bool more = (k0 + 32) < DM_;
#ifdef USE_TDM
    if (wave == 0) {
      if (more) {
        tdm_load_2d(xblk + k0 + 32, As[p ^ 1], 32, 64, DM_, 3, 3);
        tdm_load_2d(PTh  + k0 + 32, Bs[p ^ 1], 32, 32, DM_, 3, 3);
        tdm_wait2();
      } else {
        tdm_wait0();
      }
    }
#else
    {
      const int rr = tid >> 2, seg = tid & 3;        // A: 64x32, 8 elems/thread
      *(uint4*)(As[p] + rr * LDA_S + seg * 8) =
          *(const uint4*)(xblk + (size_t)rr * DM_ + k0 + seg * 8);
      if (tid < 128)                                  // B: 32x32
        *(uint4*)(Bs[p] + rr * LDA_S + seg * 8) =
            *(const uint4*)(PTh + (size_t)rr * DM_ + k0 + seg * 8);
    }
#endif
    __syncthreads();
    v16bf a  = load_frag_a(As[p] + mi1 * 16 * LDA_S, LDA_S);
    v16bf bb = load_frag_a(Bs[p] + ni1 * 16 * LDA_S, LDA_S);
    acc = wmma_bf16(a, bb, acc);
    __syncthreads();
    p ^= 1;
  }
  // Spill tmp (bf16) to LDS; stage WT (DH x R, contiguous) into Bs[0].
  {
    const int col = lane & 15, rb = (lane >> 4) ? 8 : 0;
#pragma unroll
    for (int i = 0; i < 8; ++i)
      Ts[(mi1 * 16 + rb + i) * LDA_S + ni1 * 16 + col] = (__bf16)acc[i];
  }
#ifdef USE_TDM
  if (wave == 0) { tdm_load_2d(WTh, Bs[0], 32, 64, 32, 3, 3); tdm_wait0(); }
#else
  {
    const int rr = tid >> 2, seg = tid & 3;           // 64x32
    *(uint4*)(Bs[0] + rr * LDA_S + seg * 8) =
        *(const uint4*)(WTh + (size_t)rr * R_ + seg * 8);
  }
#endif
  __syncthreads();

  // Stage 2: out = tmp @ Vw + bias ; 16 tiles, 2 per wave, single WMMA.
  __bf16* outblk = Out + ((size_t)bh * M_ + m0) * DH_;
  const int mi0 = wave >> 2, ni = wave & 3;
#pragma unroll
  for (int t = 0; t < 2; ++t) {
    const int mi = mi0 + 2 * t;
    v16bf a  = load_frag_a(Ts + mi * 16 * LDA_S, LDA_S);
    v16bf bb = load_frag_a(Bs[0] + ni * 16 * LDA_S, LDA_S);
    v8f c = {};
    c = wmma_bf16(a, bb, c);
    const int col = lane & 15, rb = (lane >> 4) ? 8 : 0;
    const int cc = ni * 16 + col;
    const float bias = bs[h * DH_ + cc];
#pragma unroll
    for (int i = 0; i < 8; ++i)
      outblk[(size_t)(mi * 16 + rb + i) * DH_ + cc] = (__bf16)((c[i] + bias) * qscale);
  }
}

// ------------------------ Flash attention kernel ----------------------------
__global__ __launch_bounds__(256) void attention_kernel(
    const __bf16* __restrict__ Qb, const __bf16* __restrict__ Kb,
    const __bf16* __restrict__ Vb, const float* __restrict__ mask,
    __bf16* __restrict__ attn) {
  const int mt = blockIdx.x;
  const int bh = blockIdx.y;
  const int b = bh / H_;
  const int tid = threadIdx.x, wave = tid >> 5, lane = tid & 31;

  __shared__ __attribute__((aligned(16))) __bf16 Qs[64 * QS_LD];
  __shared__ __attribute__((aligned(16))) __bf16 Ks[64 * QS_LD];
  __shared__ __attribute__((aligned(16))) __bf16 Ps[64 * QS_LD];
  __shared__ __attribute__((aligned(16))) __bf16 Vt[64 * QS_LD];  // transposed V
  __shared__ float Ss[64 * SS_LD];
  __shared__ float m_s[64], l_s[64], al_s[64];

  const size_t qbase = ((size_t)bh * M_ + (size_t)mt * 64) * DH_;
#ifdef USE_TDM
  if (wave == 0) { tdm_load_2d(Qb + qbase, Qs, 64, 64, 64, 4, 3); tdm_wait0(); }
#else
#pragma unroll
  for (int j = 0; j < 2; ++j) {
    const int c = tid + j * 256;
    const int rr = c >> 3, seg = c & 7;
    *(uint4*)(Qs + rr * QS_LD + seg * 8) =
        *(const uint4*)(Qb + qbase + (size_t)rr * DH_ + seg * 8);
  }
#endif
  if (tid < 64) { m_s[tid] = -3.0e38f; l_s[tid] = 0.0f; }
  __syncthreads();

  const int mi0 = wave >> 2, ni = wave & 3;
  v8f o0 = {}, o1 = {};

  for (int kt = 0; kt < M_ / 64; ++kt) {
    const size_t kbase = ((size_t)bh * M_ + (size_t)kt * 64) * DH_;
#ifdef USE_TDM
    if (wave == 0) { tdm_load_2d(Kb + kbase, Ks, 64, 64, 64, 4, 3); tdm_wait0(); }
#else
#pragma unroll
    for (int j = 0; j < 2; ++j) {
      const int c = tid + j * 256;
      const int rr = c >> 3, seg = c & 7;
      *(uint4*)(Ks + rr * QS_LD + seg * 8) =
          *(const uint4*)(Kb + kbase + (size_t)rr * DH_ + seg * 8);
    }
#endif
    // V: coalesced 16B loads, transposed scatter (row stride 36 dwords).
#pragma unroll
    for (int j = 0; j < 2; ++j) {
      const int c = tid + j * 256;
      const int rr = c >> 3, seg = c & 7;    // rr = key, seg*8.. = e
      union { uint4 v; unsigned short u[8]; } dd;
      dd.v = *(const uint4*)(Vb + kbase + (size_t)rr * DH_ + seg * 8);
#pragma unroll
      for (int u = 0; u < 8; ++u)
        Vt[(seg * 8 + u) * QS_LD + rr] = u16_as_bf16(dd.u[u]);
    }
    __syncthreads();

    // S = Q @ K^T (K^T B-fragment == A-fragment of key-major Ks).
#pragma unroll
    for (int t = 0; t < 2; ++t) {
      const int mi = mi0 + 2 * t;
      v8f s = {};
#pragma unroll
      for (int kk = 0; kk < 2; ++kk) {
        v16bf a  = load_frag_a(Qs + mi * 16 * QS_LD + kk * 32, QS_LD);
        v16bf bb = load_frag_a(Ks + ni * 16 * QS_LD + kk * 32, QS_LD);
        s = wmma_bf16(a, bb, s);
      }
      const int col = lane & 15, rb = (lane >> 4) ? 8 : 0;
#pragma unroll
      for (int i = 0; i < 8; ++i)
        Ss[(mi * 16 + rb + i) * SS_LD + ni * 16 + col] = s[i];
    }
    __syncthreads();

    // Online softmax: 4 lanes per row, 16 keys each.
    {
      const int row = tid >> 2, part = tid & 3;
      const float* mrow = mask + (size_t)b * M_ + (size_t)kt * 64;
      float sv[16];
      float mx = -3.0e38f;
#pragma unroll
      for (int j = 0; j < 16; ++j) {
        const int c = part * 16 + j;
        sv[j] = Ss[row * SS_LD + c] + mrow[c];
        mx = fmaxf(mx, sv[j]);
      }
      mx = fmaxf(mx, __shfl_xor(mx, 1, 32));
      mx = fmaxf(mx, __shfl_xor(mx, 2, 32));
      const float m_old = m_s[row];
      const float m_new = fmaxf(m_old, mx);
      const float alpha = __expf(m_old - m_new);
      float psum = 0.0f;
#pragma unroll
      for (int j = 0; j < 16; ++j) {
        const float p = __expf(sv[j] - m_new);
        psum += p;
        Ps[row * QS_LD + part * 16 + j] = (__bf16)p;
      }
      psum += __shfl_xor(psum, 1, 32);
      psum += __shfl_xor(psum, 2, 32);
      if (part == 0) {
        m_s[row]  = m_new;
        l_s[row]  = l_s[row] * alpha + psum;
        al_s[row] = alpha;
      }
    }
    __syncthreads();

    // Rescale O, accumulate O += P @ V.
#pragma unroll
    for (int t = 0; t < 2; ++t) {
      const int mi = mi0 + 2 * t;
      v8f& o = t ? o1 : o0;
      const int rb = (lane >> 4) ? 8 : 0;
#pragma unroll
      for (int i = 0; i < 8; ++i) o[i] *= al_s[mi * 16 + rb + i];
#pragma unroll
      for (int kk = 0; kk < 2; ++kk) {
        v16bf a  = load_frag_a(Ps + mi * 16 * QS_LD + kk * 32, QS_LD);
        v16bf bb = load_frag_a(Vt + ni * 16 * QS_LD + kk * 32, QS_LD);
        o = wmma_bf16(a, bb, o);
      }
    }
    __syncthreads();
  }

  // Epilogue: divide by row sum, write bf16 (B, M, H*DH).
  const int h = bh % H_;
#pragma unroll
  for (int t = 0; t < 2; ++t) {
    const int mi = mi0 + 2 * t;
    v8f& o = t ? o1 : o0;
    const int col = lane & 15, rb = (lane >> 4) ? 8 : 0;
#pragma unroll
    for (int i = 0; i < 8; ++i) {
      const int rr = mi * 16 + rb + i;
      attn[((size_t)b * M_ + (size_t)mt * 64 + rr) * DM_ + h * DH_ + ni * 16 + col] =
          (__bf16)(o[i] / l_s[rr]);
    }
  }
}

// --------------------------- Generic WMMA GEMM ------------------------------
// C[MxN] = act(A[MxK] @ B[KxN] + bias). A bf16 [M][K]; Bt bf16 pre-transposed
// [N][K]. Double-buffered TDM pipeline stages both tiles; the inner loop is
// 2 TDM issues + fragment loads + 2 WMMAs.
template <typename OT, int ACT>
__global__ __launch_bounds__(256) void gemm_kernel(
    const __bf16* __restrict__ A, const __bf16* __restrict__ Bt,
    const float* __restrict__ bias, OT* __restrict__ C,
    int K, int N) {
  const int n0 = blockIdx.x * 64;
  const int m0 = blockIdx.y * 64;
  const int tid = threadIdx.x, wave = tid >> 5, lane = tid & 31;

  __shared__ __attribute__((aligned(16))) __bf16 As[2][64 * LDA_S];
  __shared__ __attribute__((aligned(16))) __bf16 Bs[2][64 * LDA_S];

  const __bf16* Ab = A  + (size_t)m0 * K;
  const __bf16* Bb = Bt + (size_t)n0 * K;
  const int mi0 = wave >> 2, ni = wave & 3;
  v8f acc0 = {}, acc1 = {};

#ifdef USE_TDM
  if (wave == 0) {
    tdm_load_2d(Ab, As[0], 32, 64, K, 3, 3);
    tdm_load_2d(Bb, Bs[0], 32, 64, K, 3, 3);
  }
#endif
  int p = 0;
  for (int k0 = 0; k0 < K; k0 += 32) {
    const bool more = (k0 + 32) < K;
#ifdef USE_TDM
    if (wave == 0) {
      if (more) {
        tdm_load_2d(Ab + k0 + 32, As[p ^ 1], 32, 64, K, 3, 3);
        tdm_load_2d(Bb + k0 + 32, Bs[p ^ 1], 32, 64, K, 3, 3);
        tdm_wait2();
      } else {
        tdm_wait0();
      }
    }
#else
    {
      const int rr = tid >> 2, seg = tid & 3;        // 8 elems/thread each
      *(uint4*)(As[p] + rr * LDA_S + seg * 8) =
          *(const uint4*)(Ab + (size_t)rr * K + k0 + seg * 8);
      *(uint4*)(Bs[p] + rr * LDA_S + seg * 8) =
          *(const uint4*)(Bb + (size_t)rr * K + k0 + seg * 8);
    }
#endif
    __syncthreads();
    v16bf a0 = load_frag_a(As[p] + mi0 * 16 * LDA_S, LDA_S);
    v16bf a1 = load_frag_a(As[p] + (mi0 + 2) * 16 * LDA_S, LDA_S);
    v16bf bb = load_frag_a(Bs[p] + ni * 16 * LDA_S, LDA_S);
    acc0 = wmma_bf16(a0, bb, acc0);
    acc1 = wmma_bf16(a1, bb, acc1);
    __syncthreads();
    p ^= 1;
  }

  const int col = lane & 15, rb = (lane >> 4) ? 8 : 0;
  const int cc = n0 + ni * 16 + col;
  const float bv = bias ? bias[cc] : 0.0f;
#pragma unroll
  for (int i = 0; i < 8; ++i) {
    float v0 = acc0[i] + bv;
    float v1 = acc1[i] + bv;
    if (ACT == 1) { v0 = gelu_exact(v0); v1 = gelu_exact(v1); }
    C[(size_t)(m0 + mi0 * 16 + rb + i) * N + cc]       = (OT)v0;
    C[(size_t)(m0 + (mi0 + 2) * 16 + rb + i) * N + cc] = (OT)v1;
  }
}

// ----------------------- Residual + LayerNorm -------------------------------
// out = LN(x + y); optional bf16 mirror for downstream GEMM A operands.
__global__ __launch_bounds__(256) void ln_residual_kernel(
    const float* __restrict__ xin, const float* __restrict__ yin,
    const float* __restrict__ g, const float* __restrict__ bet,
    float* __restrict__ out, __bf16* __restrict__ out_bf) {
  const int row = blockIdx.x;
  const int tid = threadIdx.x;
  const float* xr = xin + (size_t)row * DM_;
  const float* yr = yin + (size_t)row * DM_;
  float v[4], s = 0.0f, s2 = 0.0f;
#pragma unroll
  for (int j = 0; j < 4; ++j) {
    const int c = tid + j * 256;
    v[j] = xr[c] + yr[c];
    s += v[j];
    s2 += v[j] * v[j];
  }
#pragma unroll
  for (int off = 16; off > 0; off >>= 1) {
    s  += __shfl_xor(s,  off, 32);
    s2 += __shfl_xor(s2, off, 32);
  }
  __shared__ float rs[8], rq[8];
  if ((tid & 31) == 0) { rs[tid >> 5] = s; rq[tid >> 5] = s2; }
  __syncthreads();
  float ts = 0.0f, tq = 0.0f;
#pragma unroll
  for (int i = 0; i < 8; ++i) { ts += rs[i]; tq += rq[i]; }
  const float mu  = ts * (1.0f / DM_);
  const float var = tq * (1.0f / DM_) - mu * mu;
  const float inv = rsqrtf(var + 1e-12f);
#pragma unroll
  for (int j = 0; j < 4; ++j) {
    const int c = tid + j * 256;
    const float r = (v[j] - mu) * inv * g[c] + bet[c];
    out[(size_t)row * DM_ + c] = r;
    if (out_bf) out_bf[(size_t)row * DM_ + c] = (__bf16)r;
  }
}

// ------------------------------ Launcher ------------------------------------
extern "C" void kernel_launch(void* const* d_in, const int* in_sizes, int n_in,
                              void* d_out, int out_size, void* d_ws, size_t ws_size,
                              hipStream_t stream) {
  (void)in_sizes; (void)n_in; (void)out_size; (void)ws_size;

  const float* x    = (const float*)d_in[0];
  const float* mask = (const float*)d_in[1];
  const float* Pq = (const float*)d_in[2],  *Vq = (const float*)d_in[3],  *bq = (const float*)d_in[4];
  const float* Pk = (const float*)d_in[5],  *Vk = (const float*)d_in[6],  *bk = (const float*)d_in[7];
  const float* Pv = (const float*)d_in[8],  *Vv = (const float*)d_in[9],  *bv = (const float*)d_in[10];
  const float* Uo = (const float*)d_in[11], *Vo = (const float*)d_in[12], *bo = (const float*)d_in[13];
  const float* g1 = (const float*)d_in[14], *beta1 = (const float*)d_in[15];
  const float* U1 = (const float*)d_in[16], *V1 = (const float*)d_in[17], *b1 = (const float*)d_in[18];
  const float* U2 = (const float*)d_in[19], *V2 = (const float*)d_in[20], *b2 = (const float*)d_in[21];
  const float* g2 = (const float*)d_in[22], *beta2 = (const float*)d_in[23];

  // ---- Workspace layout (bf16 for GEMM-fed tensors, f32 for residual/LN) ---
  char* ws = (char*)d_ws;
  size_t off = 0;
  auto carve = [&](size_t bytes) { void* r = ws + off; off += (bytes + 255) & ~(size_t)255; return r; };
  __bf16* Qb    = (__bf16*)carve((size_t)B_ * H_ * M_ * DH_ * 2);   // 16 MB
  __bf16* Kb    = (__bf16*)carve((size_t)B_ * H_ * M_ * DH_ * 2);   // 16 MB
  __bf16* Vb    = (__bf16*)carve((size_t)B_ * H_ * M_ * DH_ * 2);   // 16 MB
  __bf16* attnb = (__bf16*)carve((size_t)BM_ * DM_ * 2);            // 16 MB
  float*  x1b   = (float*) carve((size_t)BM_ * DM_ * 4);            // 32 MB
  __bf16* x1h   = (__bf16*)carve((size_t)BM_ * DM_ * 2);            // 16 MB
  __bf16* Tb    = (__bf16*)carve((size_t)BM_ * RF_ * 2);            //  8 MB
  __bf16* Hb    = (__bf16*)carve((size_t)BM_ * DFF_ * 2);           // 64 MB
  float*  Yb    = (float*) carve((size_t)BM_ * DM_ * 4);            // 32 MB
  __bf16* xh    = (__bf16*)carve((size_t)BM_ * DM_ * 2);            // 16 MB
  __bf16* UoT   = (__bf16*)carve((size_t)RF_ * DM_ * 2);
  __bf16* VoT   = (__bf16*)carve((size_t)DM_ * RF_ * 2);
  __bf16* U1T   = (__bf16*)carve((size_t)RF_ * DM_ * 2);
  __bf16* V1T   = (__bf16*)carve((size_t)DFF_ * RF_ * 2);
  __bf16* U2T   = (__bf16*)carve((size_t)RF_ * DFF_ * 2);
  __bf16* V2T   = (__bf16*)carve((size_t)DM_ * RF_ * 2);
  __bf16* PTq   = (__bf16*)carve((size_t)H_ * R_ * DM_ * 2);
  __bf16* PTk   = (__bf16*)carve((size_t)H_ * R_ * DM_ * 2);
  __bf16* PTv   = (__bf16*)carve((size_t)H_ * R_ * DM_ * 2);
  __bf16* WTq   = (__bf16*)carve((size_t)H_ * DH_ * R_ * 2);
  __bf16* WTk   = (__bf16*)carve((size_t)H_ * DH_ * R_ * 2);
  __bf16* WTv   = (__bf16*)carve((size_t)H_ * DH_ * R_ * 2);

  const dim3 blk(256);

  // 0) One-time conversions: x -> bf16; weights -> bf16, transposed [N][K].
  convert_kernel<<<dim3((BM_ * DM_) / 1024), blk, 0, stream>>>(x, xh);
  transpose_convert_kernel<<<dim3(RF_ / 32, DM_ / 32, 1),  blk, 0, stream>>>(Uo, UoT, DM_, RF_, 0, 0);
  transpose_convert_kernel<<<dim3(DM_ / 32, RF_ / 32, 1),  blk, 0, stream>>>(Vo, VoT, RF_, DM_, 0, 0);
  transpose_convert_kernel<<<dim3(RF_ / 32, DM_ / 32, 1),  blk, 0, stream>>>(U1, U1T, DM_, RF_, 0, 0);
  transpose_convert_kernel<<<dim3(DFF_ / 32, RF_ / 32, 1), blk, 0, stream>>>(V1, V1T, RF_, DFF_, 0, 0);
  transpose_convert_kernel<<<dim3(RF_ / 32, DFF_ / 32, 1), blk, 0, stream>>>(U2, U2T, DFF_, RF_, 0, 0);
  transpose_convert_kernel<<<dim3(DM_ / 32, RF_ / 32, 1),  blk, 0, stream>>>(V2, V2T, RF_, DM_, 0, 0);
  transpose_convert_kernel<<<dim3(1, DM_ / 32, H_), blk, 0, stream>>>(Pq, PTq, DM_, R_, (long)DM_ * R_, (long)R_ * DM_);
  transpose_convert_kernel<<<dim3(1, DM_ / 32, H_), blk, 0, stream>>>(Pk, PTk, DM_, R_, (long)DM_ * R_, (long)R_ * DM_);
  transpose_convert_kernel<<<dim3(1, DM_ / 32, H_), blk, 0, stream>>>(Pv, PTv, DM_, R_, (long)DM_ * R_, (long)R_ * DM_);
  transpose_convert_kernel<<<dim3(DH_ / 32, 1, H_), blk, 0, stream>>>(Vq, WTq, R_, DH_, (long)R_ * DH_, (long)DH_ * R_);
  transpose_convert_kernel<<<dim3(DH_ / 32, 1, H_), blk, 0, stream>>>(Vk, WTk, R_, DH_, (long)R_ * DH_, (long)DH_ * R_);
  transpose_convert_kernel<<<dim3(DH_ / 32, 1, H_), blk, 0, stream>>>(Vv, WTv, R_, DH_, (long)R_ * DH_, (long)DH_ * R_);

  // 1) Q/K/V low-rank projections (Q pre-scaled by 1/sqrt(DH)).
  qkv_kernel<<<dim3(M_ / 64, B_ * H_, 3), blk, 0, stream>>>(
      xh, PTq, PTk, PTv, WTq, WTk, WTv, bq, bk, bv, Qb, Kb, Vb);

  // 2) Flash attention (TDM-staged tiles) -> bf16 (B, M, DM).
  attention_kernel<<<dim3(M_ / 64, B_ * H_), blk, 0, stream>>>(
      Qb, Kb, Vb, mask, attnb);

  // 3) Out-projection (low-rank) + residual + LN1 (emits f32 + bf16 x1).
  gemm_kernel<__bf16, 0><<<dim3(RF_ / 64, BM_ / 64), blk, 0, stream>>>(attnb, UoT, nullptr, Tb, DM_, RF_);
  gemm_kernel<float , 0><<<dim3(DM_ / 64, BM_ / 64), blk, 0, stream>>>(Tb, VoT, bo, Yb, RF_, DM_);
  ln_residual_kernel<<<dim3(BM_), blk, 0, stream>>>(x, Yb, g1, beta1, x1b, x1h);

  // 4) FFN: GELU((x1@U1)@V1 + b1) @ U2 @ V2 + b2, residual + LN2.
  gemm_kernel<__bf16, 0><<<dim3(RF_ / 64, BM_ / 64), blk, 0, stream>>>(x1h, U1T, nullptr, Tb, DM_, RF_);
  gemm_kernel<__bf16, 1><<<dim3(DFF_ / 64, BM_ / 64), blk, 0, stream>>>(Tb, V1T, b1, Hb, RF_, DFF_);
  gemm_kernel<__bf16, 0><<<dim3(RF_ / 64, BM_ / 64), blk, 0, stream>>>(Hb, U2T, nullptr, Tb, DFF_, RF_);
  gemm_kernel<float , 0><<<dim3(DM_ / 64, BM_ / 64), blk, 0, stream>>>(Tb, V2T, b2, Yb, RF_, DM_);
  ln_residual_kernel<<<dim3(BM_), blk, 0, stream>>>(x1b, Yb, g2, beta2, (float*)d_out, nullptr);
}
